// SparseMOE_7541962572503
// MI455X (gfx1250) — matrix operand order repaired and checked by reading, not compiled, not run on hardware
//
#include <hip/hip_runtime.h>
#include <hip/hip_bf16.h>

#define D_MODEL 1024
#define D_HIDDEN 4096
#define NE 8
#define NT 4096          // B*S tokens
#define NSLOT (NT*2)     // top-2 -> 2T slots total

typedef __attribute__((ext_vector_type(16))) __bf16 v16bf;
typedef __attribute__((ext_vector_type(8)))  float  v8f;
typedef __attribute__((ext_vector_type(4)))  int    v4i;

#if defined(__has_builtin)
#if __has_builtin(__builtin_amdgcn_global_load_async_to_lds_b128)
#define USE_ASYNC_COPY 1
#endif
#endif
#ifndef USE_ASYNC_COPY
#define USE_ASYNC_COPY 0
#endif

#if USE_ASYNC_COPY
#if __has_builtin(__builtin_amdgcn_s_wait_asynccnt)
#define WAIT_ASYNC(n) __builtin_amdgcn_s_wait_asynccnt(n)
#else
#define WAIT_ASYNC(n) asm volatile("s_wait_asynccnt %0" :: "i"(n) : "memory")
#endif
#else
#define WAIT_ASYNC(n) ((void)0)
#endif

// per-lane 16-byte copy global -> LDS (async on CDNA5, sync fallback)
__device__ __forceinline__ void cp_b128(const unsigned short* g, __bf16* l) {
#if USE_ASYNC_COPY
  __builtin_amdgcn_global_load_async_to_lds_b128((v4i*)g, (v4i*)l, 0, 0);
#else
  uint4 v = *(const uint4*)g;
  *(uint4*)l = v;
#endif
}

__device__ __forceinline__ unsigned short f2bf(float f) {
  __bf16 b = (__bf16)f;
  return __builtin_bit_cast(unsigned short, b);
}

// ---------------------------------------------------------------------------
// 0. zero the per-expert counters / prob sums
// ---------------------------------------------------------------------------
__global__ void moe_zero(int* cnt, float* psum) {
  if (threadIdx.x < NE) { cnt[threadIdx.x] = 0; psum[threadIdx.x] = 0.f; }
}

// ---------------------------------------------------------------------------
// 1. convert x fp32 -> bf16 (row-major, K-contiguous for WMMA A frags)
// ---------------------------------------------------------------------------
__global__ void moe_convert_x(const float* __restrict__ x, unsigned short* __restrict__ xb) {
  size_t i = ((size_t)blockIdx.x * 256 + threadIdx.x) * 4;
  float4 v = *(const float4*)(x + i);
  unsigned short o[4] = { f2bf(v.x), f2bf(v.y), f2bf(v.z), f2bf(v.w) };
  *(uint2*)(xb + i) = *(const uint2*)o;
}

// ---------------------------------------------------------------------------
// 2. transpose+convert a weight slab: src fp32 [R][C] -> dst bf16 [C][R]
//    grid: (C/64, R/64, NE)
// ---------------------------------------------------------------------------
__global__ void moe_transpose_bf16(const float* __restrict__ src, unsigned short* __restrict__ dst,
                                   int R, int C) {
  __shared__ float tile[64][65];
  const float* s = src + (size_t)blockIdx.z * R * C;
  unsigned short* d = dst + (size_t)blockIdx.z * R * C;
  int rb = blockIdx.y * 64, cb = blockIdx.x * 64;
  int tx = threadIdx.x & 63, ty = threadIdx.x >> 6;   // 4 rows per pass
  #pragma unroll
  for (int i = 0; i < 16; i++) {
    int r = ty + i * 4;
    tile[r][tx] = s[(size_t)(rb + r) * C + cb + tx];
  }
  __syncthreads();
  #pragma unroll
  for (int i = 0; i < 16; i++) {
    int r = ty + i * 4;
    d[(size_t)(cb + r) * R + rb + tx] = f2bf(tile[tx][r]);
  }
}

// ---------------------------------------------------------------------------
// 3. router: logits -> softmax -> top-2 -> slot assignment + P_i sums
// ---------------------------------------------------------------------------
__global__ void moe_router(const float* __restrict__ x, const float* __restrict__ rw,
                           const float* __restrict__ rb,
                           int* __restrict__ cnt, float* __restrict__ psum,
                           int* __restrict__ list,
                           int* __restrict__ tk_e, int* __restrict__ tk_i,
                           float* __restrict__ tk_w) {
  __shared__ float sp[NE];
  if (threadIdx.x < NE) sp[threadIdx.x] = 0.f;
  __syncthreads();
  int t = blockIdx.x * 256 + threadIdx.x;
  float lg[NE];
  #pragma unroll
  for (int e = 0; e < NE; e++) lg[e] = rb[e];
  const float* xr = x + (size_t)t * D_MODEL;
  for (int k = 0; k < D_MODEL; k++) {
    float xv = xr[k];
    const float* r = rw + (size_t)k * NE;
    #pragma unroll
    for (int e = 0; e < NE; e++) lg[e] = fmaf(xv, r[e], lg[e]);
  }
  float mx = lg[0];
  #pragma unroll
  for (int e = 1; e < NE; e++) mx = fmaxf(mx, lg[e]);
  float p[NE], s = 0.f;
  #pragma unroll
  for (int e = 0; e < NE; e++) { p[e] = __expf(lg[e] - mx); s += p[e]; }
  float inv = 1.f / s;
  #pragma unroll
  for (int e = 0; e < NE; e++) p[e] *= inv;
  int e0 = 0;
  #pragma unroll
  for (int e = 1; e < NE; e++) if (p[e] > p[e0]) e0 = e;
  int e1 = (e0 == 0) ? 1 : 0;
  #pragma unroll
  for (int e = 0; e < NE; e++) if (e != e0 && p[e] > p[e1]) e1 = e;
  int es[2] = { e0, e1 };
  #pragma unroll
  for (int k = 0; k < 2; k++) {
    int e = es[k];
    int i = atomicAdd(&cnt[e], 1);
    list[e * NT + i] = t;
    tk_e[t * 2 + k] = e;
    tk_i[t * 2 + k] = i;
    tk_w[t * 2 + k] = p[e];
  }
  #pragma unroll
  for (int e = 0; e < NE; e++) atomicAdd(&sp[e], p[e]);
  __syncthreads();
  if (threadIdx.x < NE) atomicAdd(&psum[threadIdx.x], sp[threadIdx.x]);
}

// ---------------------------------------------------------------------------
// 4. segment scan over counts + load-balance loss
// ---------------------------------------------------------------------------
__global__ void moe_seg_loss(const int* __restrict__ cnt, const float* __restrict__ psum,
                             int* __restrict__ seg, float* __restrict__ out_loss) {
  if (threadIdx.x == 0) {
    int s = 0;
    float l = 0.f;
    for (int e = 0; e < NE; e++) {
      seg[e] = s; s += cnt[e];
      l += ((float)cnt[e] / (float)NT) * (psum[e] / (float)NT);
    }
    seg[NE] = s;
    out_loss[0] = 0.01f * (float)NE * l;
  }
}

// ---------------------------------------------------------------------------
// 5. GEMM1: h = (Xg@w1 + b1) * silu(Xg@w2 + b2), bf16 WMMA, f32 accum
//    block tile 128(M) x 64(N), 8 waves, double-buffered async LDS fills
//    per-wave async batch = 4 (2xA + B1 + B2) -> s_wait_asynccnt 4
//    grid: (NE * NT/128, D_HIDDEN/64)
// ---------------------------------------------------------------------------
__global__ __launch_bounds__(256, 1) void moe_ffn1(
    const unsigned short* __restrict__ xb,
    const unsigned short* __restrict__ w1t,   // [E][H][D] bf16, K-contig
    const unsigned short* __restrict__ w2t,
    const float* __restrict__ b1, const float* __restrict__ b2,
    const int* __restrict__ cnt, const int* __restrict__ seg,
    const int* __restrict__ list,
    unsigned short* __restrict__ hbuf) {      // [NSLOT][H] bf16
  __shared__ alignas(32) __bf16 lsA [2][8][32][16];   // 16 KB
  __shared__ alignas(32) __bf16 lsB1[2][4][32][16];   //  8 KB
  __shared__ alignas(32) __bf16 lsB2[2][4][32][16];   //  8 KB

  const int e = blockIdx.x >> 5;
  const int mbase = (blockIdx.x & 31) * 128;
  const int c = cnt[e];
  if (mbase >= c) return;
  const int nbase = blockIdx.y * 64;
  const int tid = threadIdx.x, lane = tid & 31, wave = tid >> 5;

  // A source: row clamped to last valid token (finite data, stores are guarded)
  const int arow = wave * 16 + (lane & 15);
  const int mc = min(mbase + arow, c - 1);
  const int tok = list[e * NT + mc];
  const unsigned short* asrc = xb + (size_t)tok * D_MODEL + ((lane < 16) ? 0 : 8);

  // B fill: 16 B per thread, all 256 threads, for each of B1/B2
  const int bnt = tid >> 6;          // 0..3 n-tile
  const int bsub = tid & 63;
  const int blane = bsub >> 1;       // frag lane 0..31
  const int bhalf = bsub & 1;        // which 8-half chunk
  const int bn = nbase + bnt * 16 + (blane & 15);
  const int bko = ((blane < 16) ? 0 : 8) + bhalf * 16;
  const unsigned short* b1src = w1t + ((size_t)e * D_HIDDEN + bn) * D_MODEL + bko;
  const unsigned short* b2src = w2t + ((size_t)e * D_HIDDEN + bn) * D_MODEL + bko;

  auto issue = [&](int buf, int kb) {
    cp_b128(asrc + kb,      &lsA[buf][wave][lane][0]);
    cp_b128(asrc + kb + 16, &lsA[buf][wave][lane][8]);
    cp_b128(b1src + kb, &lsB1[buf][bnt][blane][bhalf * 8]);
    cp_b128(b2src + kb, &lsB2[buf][bnt][blane][bhalf * 8]);
  };

  v8f acc1[4] = {}, acc2[4] = {};

  issue(0, 0);
  for (int kb = 0; kb < D_MODEL; kb += 32) {
    const int buf = (kb >> 5) & 1;
    if (kb + 32 < D_MODEL) {
      issue(buf ^ 1, kb + 32);   // overlap next fill with this tile's WMMAs
      WAIT_ASYNC(4);             // current tile's batch complete
    } else {
      WAIT_ASYNC(0);
    }
    __syncthreads();

    v16bf a = *(const v16bf*)&lsA[buf][wave][lane][0];
    #pragma unroll
    for (int nt = 0; nt < 4; nt++) {
      v16bf fb1 = *(const v16bf*)&lsB1[buf][nt][lane][0];
      acc1[nt] = __builtin_amdgcn_wmma_f32_16x16x32_bf16(
          false, a, false, fb1, (short)0, acc1[nt], false, false);
      v16bf fb2 = *(const v16bf*)&lsB2[buf][nt][lane][0];
      acc2[nt] = __builtin_amdgcn_wmma_f32_16x16x32_bf16(
          false, a, false, fb2, (short)0, acc2[nt], false, false);
    }
    __syncthreads();             // all reads of `buf` done before it is refilled
  }

  const int segb = seg[e];
  #pragma unroll
  for (int nt = 0; nt < 4; nt++) {
    int n = nbase + nt * 16 + (lane & 15);
    float bb1 = b1[e * D_HIDDEN + n];
    float bb2 = b2[e * D_HIDDEN + n];
    #pragma unroll
    for (int v = 0; v < 8; v++) {
      int mm = mbase + wave * 16 + v + ((lane < 16) ? 0 : 8);
      if (mm < c) {
        float h1 = acc1[nt][v] + bb1;
        float h2 = acc2[nt][v] + bb2;
        float hv = h1 * (h2 / (1.f + __expf(-h2)));   // h1 * silu(h2)
        hbuf[(size_t)(segb + mm) * D_HIDDEN + n] = f2bf(hv);
      }
    }
  }
}

// ---------------------------------------------------------------------------
// 6. GEMM2: outs = h @ wo + bo (per slot), bf16 WMMA, f32 out
//    per-wave async batch = 3 (2xA + B) -> s_wait_asynccnt 3
//    grid: (NE * NT/128, D_MODEL/64)
// ---------------------------------------------------------------------------
__global__ __launch_bounds__(256, 1) void moe_ffn2(
    const unsigned short* __restrict__ hbuf,
    const unsigned short* __restrict__ wot,   // [E][D][H] bf16, K-contig
    const float* __restrict__ bo,
    const int* __restrict__ cnt, const int* __restrict__ seg,
    float* __restrict__ outs) {               // [NSLOT][D] f32
  __shared__ alignas(32) __bf16 lsA[2][8][32][16];    // 16 KB
  __shared__ alignas(32) __bf16 lsB[2][4][32][16];    //  8 KB

  const int e = blockIdx.x >> 5;
  const int mbase = (blockIdx.x & 31) * 128;
  const int c = cnt[e];
  if (mbase >= c) return;
  const int nbase = blockIdx.y * 64;
  const int tid = threadIdx.x, lane = tid & 31, wave = tid >> 5;
  const int segb = seg[e];

  const int arow = wave * 16 + (lane & 15);
  const int mc = min(mbase + arow, c - 1);
  const unsigned short* asrc = hbuf + (size_t)(segb + mc) * D_HIDDEN + ((lane < 16) ? 0 : 8);

  const int bnt = tid >> 6;
  const int bsub = tid & 63;
  const int blane = bsub >> 1;
  const int bhalf = bsub & 1;
  const int bn = nbase + bnt * 16 + (blane & 15);
  const int bko = ((blane < 16) ? 0 : 8) + bhalf * 16;
  const unsigned short* bsrc = wot + ((size_t)e * D_MODEL + bn) * D_HIDDEN + bko;

  auto issue = [&](int buf, int kb) {
    cp_b128(asrc + kb,      &lsA[buf][wave][lane][0]);
    cp_b128(asrc + kb + 16, &lsA[buf][wave][lane][8]);
    cp_b128(bsrc + kb, &lsB[buf][bnt][blane][bhalf * 8]);
  };

  v8f acc[4] = {};

  issue(0, 0);
  for (int kb = 0; kb < D_HIDDEN; kb += 32) {
    const int buf = (kb >> 5) & 1;
    if (kb + 32 < D_HIDDEN) {
      issue(buf ^ 1, kb + 32);
      WAIT_ASYNC(3);
    } else {
      WAIT_ASYNC(0);
    }
    __syncthreads();

    v16bf a = *(const v16bf*)&lsA[buf][wave][lane][0];
    #pragma unroll
    for (int nt = 0; nt < 4; nt++) {
      v16bf fb = *(const v16bf*)&lsB[buf][nt][lane][0];
      acc[nt] = __builtin_amdgcn_wmma_f32_16x16x32_bf16(
          false, a, false, fb, (short)0, acc[nt], false, false);
    }
    __syncthreads();
  }

  #pragma unroll
  for (int nt = 0; nt < 4; nt++) {
    int n = nbase + nt * 16 + (lane & 15);
    float bb = bo[e * D_MODEL + n];
    #pragma unroll
    for (int v = 0; v < 8; v++) {
      int mm = mbase + wave * 16 + v + ((lane < 16) ? 0 : 8);
      if (mm < c)
        outs[(size_t)(segb + mm) * D_MODEL + n] = acc[nt][v] + bb;
    }
  }
}

// ---------------------------------------------------------------------------
// 7. combine: out[t] = sum_k w_k * outs[slot_k]
// ---------------------------------------------------------------------------
__global__ void moe_combine(const float* __restrict__ outs,
                            const int* __restrict__ tk_e, const int* __restrict__ tk_i,
                            const float* __restrict__ tk_w, const int* __restrict__ seg,
                            float* __restrict__ out) {
  size_t idx = (size_t)blockIdx.x * 256 + threadIdx.x;   // over NT * D/4
  int t = (int)(idx / (D_MODEL / 4));
  int d4 = (int)(idx % (D_MODEL / 4)) * 4;
  float4 r = {0.f, 0.f, 0.f, 0.f};
  #pragma unroll
  for (int k = 0; k < 2; k++) {
    int e = tk_e[t * 2 + k];
    int i = tk_i[t * 2 + k];
    float w = tk_w[t * 2 + k];
    const float4 v = *(const float4*)(outs + (size_t)(seg[e] + i) * D_MODEL + d4);
    r.x += w * v.x; r.y += w * v.y; r.z += w * v.z; r.w += w * v.w;
  }
  *(float4*)(out + (size_t)t * D_MODEL + d4) = r;
}

// ---------------------------------------------------------------------------
extern "C" void kernel_launch(void* const* d_in, const int* in_sizes, int n_in,
                              void* d_out, int out_size, void* d_ws, size_t ws_size,
                              hipStream_t stream) {
  const float* x  = (const float*)d_in[0];
  const float* rw = (const float*)d_in[1];
  const float* rb = (const float*)d_in[2];
  const float* w1 = (const float*)d_in[3];
  const float* b1 = (const float*)d_in[4];
  const float* w2 = (const float*)d_in[5];
  const float* b2 = (const float*)d_in[6];
  const float* wo = (const float*)d_in[7];
  const float* bo = (const float*)d_in[8];
  float* out = (float*)d_out;

  char* ws = (char*)d_ws;
  size_t off = 0;
  auto carve = [&](size_t bytes) -> char* {
    char* p = ws + off;
    off += (bytes + 255) & ~(size_t)255;
    return p;
  };
  unsigned short* xb   = (unsigned short*)carve((size_t)NT * D_MODEL * 2);
  unsigned short* w1t  = (unsigned short*)carve((size_t)NE * D_HIDDEN * D_MODEL * 2);
  unsigned short* w2t  = (unsigned short*)carve((size_t)NE * D_HIDDEN * D_MODEL * 2);
  unsigned short* wot  = (unsigned short*)carve((size_t)NE * D_MODEL * D_HIDDEN * 2);
  unsigned short* hbuf = (unsigned short*)carve((size_t)NSLOT * D_HIDDEN * 2);
  float* outs          = (float*)carve((size_t)NSLOT * D_MODEL * 4);
  int*   cnt           = (int*)carve(NE * 4);
  float* psum          = (float*)carve(NE * 4);
  int*   seg           = (int*)carve((NE + 1) * 4);
  int*   list          = (int*)carve((size_t)NE * NT * 4);
  int*   tk_e          = (int*)carve((size_t)NT * 2 * 4);
  int*   tk_i          = (int*)carve((size_t)NT * 2 * 4);
  float* tk_w          = (float*)carve((size_t)NT * 2 * 4);

  moe_zero<<<1, 32, 0, stream>>>(cnt, psum);

  moe_convert_x<<<(NT * D_MODEL / 4) / 256, 256, 0, stream>>>(x, xb);

  // w1 [E][D][H] -> w1t [E][H][D];  w2 same;  wo [E][H][D] -> wot [E][D][H]
  moe_transpose_bf16<<<dim3(D_HIDDEN / 64, D_MODEL / 64, NE), 256, 0, stream>>>(w1, w1t, D_MODEL, D_HIDDEN);
  moe_transpose_bf16<<<dim3(D_HIDDEN / 64, D_MODEL / 64, NE), 256, 0, stream>>>(w2, w2t, D_MODEL, D_HIDDEN);
  moe_transpose_bf16<<<dim3(D_MODEL / 64, D_HIDDEN / 64, NE), 256, 0, stream>>>(wo, wot, D_HIDDEN, D_MODEL);

  moe_router<<<NT / 256, 256, 0, stream>>>(x, rw, rb, cnt, psum, list, tk_e, tk_i, tk_w);

  moe_seg_loss<<<1, 32, 0, stream>>>(cnt, psum, seg, out + (size_t)NT * D_MODEL);

  moe_ffn1<<<dim3(NE * (NT / 128), D_HIDDEN / 64), 256, 0, stream>>>(
      xb, w1t, w2t, b1, b2, cnt, seg, list, hbuf);

  moe_ffn2<<<dim3(NE * (NT / 128), D_MODEL / 64), 256, 0, stream>>>(
      hbuf, wot, bo, cnt, seg, outs);

  moe_combine<<<(NT * (D_MODEL / 4)) / 256, 256, 0, stream>>>(
      outs, tk_e, tk_i, tk_w, seg, out);
}
